// EncoderLayer_26731876450515
// MI455X (gfx1250) — matrix operand order repaired and checked by reading, not compiled, not run on hardware
//
#include <hip/hip_runtime.h>

typedef __bf16 bf16;
typedef __attribute__((ext_vector_type(16))) bf16 v16bf;
typedef __attribute__((ext_vector_type(8)))  bf16 bf16x8;
typedef __attribute__((ext_vector_type(8)))  float v8f;

__device__ __forceinline__ v8f wmma_bf16f32(v16bf a, v16bf b, v8f c) {
  // D = A(16x32 bf16) * B(32x16 bf16) + C(16x16 f32)
  return __builtin_amdgcn_wmma_f32_16x16x32_bf16(false, a, false, b, (short)0, c,
                                                 false, false);
}

__device__ __forceinline__ v16bf pack16(bf16x8 lo, bf16x8 hi) {
  v16bf r;
#pragma unroll
  for (int j = 0; j < 8; ++j) { r[j] = lo[j]; r[j + 8] = hi[j]; }
  return r;
}

// CDNA5 async global->LDS DMA (VGLOBAL op 98, tracked with ASYNCcnt).
// VDST = per-lane LDS byte address; generic __shared__ pointer low 32 bits are
// the LDS address (flat-aperture rule). GV addressing mode (saddr = off).
__device__ __forceinline__ void async_load_b128(const void* gaddr, void* lds) {
  unsigned l = (unsigned)(unsigned long long)lds;
  asm volatile("global_load_async_to_lds_b128 %0, %1, off"
               :: "v"(l), "v"(gaddr) : "memory");
}
__device__ __forceinline__ void wait_async0() {
  asm volatile("s_wait_asynccnt 0x0" ::: "memory");
}

// ---------------------------------------------------------------------------
// f32 -> bf16 conversion (plain)
// ---------------------------------------------------------------------------
__global__ __launch_bounds__(256) void cvt_kernel(const float* __restrict__ src,
                                                  bf16* __restrict__ dst, int n) {
  int i = blockIdx.x * 256 + threadIdx.x;
  if (i < n) dst[i] = (bf16)src[i];
}

// ---------------------------------------------------------------------------
// f32 [R][C] -> bf16 transposed [C][R], 32x32 LDS tile, coalesced both sides
// ---------------------------------------------------------------------------
__global__ __launch_bounds__(256) void cvt_t_kernel(const float* __restrict__ in,
                                                    bf16* __restrict__ out,
                                                    int R, int C) {
  __shared__ float tile[32][33];
  const int c0 = blockIdx.x * 32, r0 = blockIdx.y * 32;
  const int tid = threadIdx.x;
#pragma unroll
  for (int i = 0; i < 4; ++i) {
    int idx = tid + i * 256;
    int r = idx >> 5, c = idx & 31;
    tile[r][c] = in[(size_t)(r0 + r) * C + c0 + c];
  }
  __syncthreads();
#pragma unroll
  for (int i = 0; i < 4; ++i) {
    int idx = tid + i * 256;
    int c = idx >> 5, r = idx & 31;
    out[(size_t)(c0 + c) * R + r0 + r] = (bf16)tile[r][c];
  }
}

// ---------------------------------------------------------------------------
// bf16 WMMA GEMM: C[M,N] = A[M,K] @ BT[N,K]^T (+bias) (optional ReLU)
// 128x128 block tile, K-step 64, 4 waves, 64x64 wave tile (2x2 wave grid).
// Per K=32 step: 4 A-frags + 4 B-frags (16 ds_load_b128) feed 16 WMMA
// (1.0 LDS loads per WMMA). Double-buffered async global->LDS staging
// overlaps the DMA of tile t+1 with the WMMA of tile t.
// A-fragment: lane m=l16, j<8 -> k=hx*8+j ; j>=8 -> k=16+hx*8+(j-8).
// B-fragment: lane n=l16, j<8 -> k=hx*8+j ; j>=8 -> k=16+hx*8+(j-8) (BT rows contiguous in k).
// C/D: vgpr r -> m = r + 8*hx, n = l16.
// storeT: write Cb transposed ([N][M]) for the V matrix.
// ---------------------------------------------------------------------------
__global__ __launch_bounds__(128) void gemm_bf16_kernel(
    const bf16* __restrict__ A, const bf16* __restrict__ BT,
    const float* __restrict__ bias,
    float* __restrict__ Cf, bf16* __restrict__ Cb,
    int M, int N, int K, int relu, int storeT) {
  __shared__ bf16 As[2][128][64];   // 2 x 16 KB, [m][k]
  __shared__ bf16 Bs[2][128][64];   // 2 x 16 KB, [n][k]
  const int tid = threadIdx.x;
  const int wave = tid >> 5, lane = tid & 31;
  const int hx = lane >> 4, l16 = lane & 15;
  const int wm = wave & 1, wn = wave >> 1;   // 2x2 wave grid, 64x64 tiles
  const int bm = blockIdx.y * 128, bn = blockIdx.x * 128;
  const int ntiles = K >> 6;

  v8f acc[4][4] = {};

  // stage tile 0 into buffer 0 (1024 16B chunks per operand, 8 per thread)
#pragma unroll
  for (int i = 0; i < 8; ++i) {
    int c = tid * 8 + i;
    int row = c >> 3, kc = (c & 7) * 8;
    async_load_b128(A + (size_t)(bm + row) * K + kc, &As[0][row][kc]);
    async_load_b128(BT + (size_t)(bn + row) * K + kc, &Bs[0][row][kc]);
  }
  wait_async0();
  __syncthreads();

  for (int kt = 0; kt < ntiles; ++kt) {
    const int cur = kt & 1;
    const int k0n = (kt + 1) << 6;
    if (kt + 1 < ntiles) {  // stage next tile while computing current
#pragma unroll
      for (int i = 0; i < 8; ++i) {
        int c = tid * 8 + i;
        int row = c >> 3, kc = (c & 7) * 8;
        async_load_b128(A + (size_t)(bm + row) * K + k0n + kc,
                        &As[cur ^ 1][row][kc]);
        async_load_b128(BT + (size_t)(bn + row) * K + k0n + kc,
                        &Bs[cur ^ 1][row][kc]);
      }
    }

#pragma unroll
    for (int kk = 0; kk < 64; kk += 32) {
      v16bf av[4], bv[4];
#pragma unroll
      for (int ms = 0; ms < 4; ++ms) {
        int row = wm * 64 + ms * 16 + l16;
        av[ms] = pack16(*(const bf16x8*)(&As[cur][row][kk + hx * 8]),
                        *(const bf16x8*)(&As[cur][row][kk + 16 + hx * 8]));
      }
#pragma unroll
      for (int ns = 0; ns < 4; ++ns) {
        int col = wn * 64 + ns * 16 + l16;
        bv[ns] = pack16(*(const bf16x8*)(&Bs[cur][col][kk + hx * 8]),
                        *(const bf16x8*)(&Bs[cur][col][kk + 16 + hx * 8]));
      }
#pragma unroll
      for (int ms = 0; ms < 4; ++ms)
#pragma unroll
        for (int ns = 0; ns < 4; ++ns)
          acc[ms][ns] = wmma_bf16f32(av[ms], bv[ns], acc[ms][ns]);
    }

    wait_async0();      // own async copies for next buffer complete
    __syncthreads();    // all waves done computing cur / staging cur^1
  }

  // epilogue: bias (+ReLU), write f32 and/or bf16 (optionally transposed)
#pragma unroll
  for (int ms = 0; ms < 4; ++ms) {
#pragma unroll
    for (int ns = 0; ns < 4; ++ns) {
      int col = bn + wn * 64 + ns * 16 + l16;
      float bb = bias ? bias[col] : 0.0f;
#pragma unroll
      for (int r = 0; r < 8; ++r) {
        int row = bm + wm * 64 + ms * 16 + r + 8 * hx;
        float v = acc[ms][ns][r] + bb;
        if (relu) v = fmaxf(v, 0.0f);
        size_t idx = storeT ? ((size_t)col * M + row) : ((size_t)row * N + col);
        if (Cf) Cf[idx] = v;
        if (Cb) Cb[idx] = (bf16)v;
      }
    }
  }
}

// ---------------------------------------------------------------------------
// Flash attention: grid = (S/128, B*H). 8 waves, each owns 16 query rows.
// K tile (32 keys x 64 dk, row-major) and V tile (from pre-transposed VT,
// [dk][key]) staged with double-buffered async copies; P (16x32) round-trips
// per-wave LDS to convert C/D layout -> A-operand layout for the P@V WMMA.
// ---------------------------------------------------------------------------
__global__ __launch_bounds__(256) void attn_kernel(
    const bf16* __restrict__ Q, const bf16* __restrict__ Km,
    const bf16* __restrict__ VT, bf16* __restrict__ ctx, int S, int Ntok) {
  const int D = 1024, DK = 64;
  __shared__ bf16 Kt[2][32][64];   // [key][dk]
  __shared__ bf16 Vt[2][64][32];   // [dk][key]
  __shared__ bf16 Pb[8][16][32];   // per-wave P staging
  const int tid = threadIdx.x;
  const int wave = tid >> 5, lane = tid & 31;
  const int hx = lane >> 4, l16 = lane & 15;
  const int bh = blockIdx.y, b = bh >> 4, h = bh & 15;
  const int q0 = blockIdx.x * 128 + wave * 16;

  // Q fragments (A-operand, 16 rows x 64 dk = 2 K-chunks) straight from global
  const bf16* qptr = Q + ((size_t)(b * S + q0 + l16)) * D + h * DK;
  v16bf qa[2];
#pragma unroll
  for (int c = 0; c < 2; ++c)
    qa[c] = pack16(*(const bf16x8*)(qptr + c * 32 + hx * 8),
                   *(const bf16x8*)(qptr + c * 32 + 16 + hx * 8));

  v8f o[4] = {};
  float mrow[8], lrow[8];
#pragma unroll
  for (int r = 0; r < 8; ++r) { mrow[r] = -1e30f; lrow[r] = 0.0f; }

  const int key_ld = tid >> 3, ch = (tid & 7) * 8;   // K tile: 256 chunks
  const int dk_ld = tid >> 2, vkc = (tid & 3) * 8;   // V tile: 256 chunks

  // stage tile 0
  async_load_b128(Km + (size_t)(b * S + key_ld) * D + h * DK + ch,
                  &Kt[0][key_ld][ch]);
  async_load_b128(VT + (size_t)(h * DK + dk_ld) * Ntok + b * S + vkc,
                  &Vt[0][dk_ld][vkc]);
  wait_async0();
  __syncthreads();

  const int ntiles = S >> 5;
  for (int kt = 0; kt < ntiles; ++kt) {
    const int cur = kt & 1;
    const int k0n = (kt + 1) << 5;
    if (kt + 1 < ntiles) {
      async_load_b128(Km + (size_t)(b * S + k0n + key_ld) * D + h * DK + ch,
                      &Kt[cur ^ 1][key_ld][ch]);
      async_load_b128(VT + (size_t)(h * DK + dk_ld) * Ntok + b * S + k0n + vkc,
                      &Vt[cur ^ 1][dk_ld][vkc]);
    }

    // scores: two 16x16 tiles (keys 0..15, 16..31), contraction over dk=64
    v8f s[2] = {};
#pragma unroll
    for (int ns = 0; ns < 2; ++ns) {
      int key = ns * 16 + l16;
#pragma unroll
      for (int c = 0; c < 2; ++c) {
        v16bf bvec = pack16(*(const bf16x8*)(&Kt[cur][key][c * 32 + hx * 8]),
                            *(const bf16x8*)(&Kt[cur][key][c * 32 + 16 + hx * 8]));
        s[ns] = wmma_bf16f32(qa[c], bvec, s[ns]);
      }
      s[ns] = s[ns] * 0.125f;  // 1/sqrt(64)
    }

    // online softmax (row = r + 8*hx, the 16 lanes of a column group)
    float mt[8];
#pragma unroll
    for (int r = 0; r < 8; ++r) mt[r] = fmaxf(s[0][r], s[1][r]);
#pragma unroll
    for (int off = 1; off < 16; off <<= 1)
#pragma unroll
      for (int r = 0; r < 8; ++r) mt[r] = fmaxf(mt[r], __shfl_xor(mt[r], off));

    float alpha[8], p0[8], p1[8], rs[8];
#pragma unroll
    for (int r = 0; r < 8; ++r) {
      float mn = fmaxf(mrow[r], mt[r]);
      alpha[r] = __expf(mrow[r] - mn);
      mrow[r] = mn;
      p0[r] = __expf(s[0][r] - mn);
      p1[r] = __expf(s[1][r] - mn);
      rs[r] = p0[r] + p1[r];
    }
#pragma unroll
    for (int off = 1; off < 16; off <<= 1)
#pragma unroll
      for (int r = 0; r < 8; ++r) rs[r] += __shfl_xor(rs[r], off);
#pragma unroll
    for (int r = 0; r < 8; ++r) lrow[r] = lrow[r] * alpha[r] + rs[r];
#pragma unroll
    for (int d = 0; d < 4; ++d)
#pragma unroll
      for (int r = 0; r < 8; ++r) o[d][r] *= alpha[r];

    // P: C-layout -> per-wave LDS -> A-layout (wave-local LDS stays in order)
#pragma unroll
    for (int r = 0; r < 8; ++r) {
      Pb[wave][r + 8 * hx][l16] = (bf16)p0[r];
      Pb[wave][r + 8 * hx][16 + l16] = (bf16)p1[r];
    }
    v16bf pa = pack16(*(const bf16x8*)(&Pb[wave][l16][hx * 8]),
                      *(const bf16x8*)(&Pb[wave][l16][16 + hx * 8]));
#pragma unroll
    for (int d = 0; d < 4; ++d) {
      int dk = d * 16 + l16;
      v16bf vb = pack16(*(const bf16x8*)(&Vt[cur][dk][hx * 8]),
                        *(const bf16x8*)(&Vt[cur][dk][16 + hx * 8]));
      o[d] = wmma_bf16f32(pa, vb, o[d]);  // contraction over 32 keys
    }

    wait_async0();
    __syncthreads();
  }

  // normalize and write context (bf16, layout [token][h*64+dk])
#pragma unroll
  for (int d = 0; d < 4; ++d)
#pragma unroll
    for (int r = 0; r < 8; ++r) {
      int row = q0 + r + 8 * hx;
      int col = h * DK + d * 16 + l16;
      ctx[((size_t)(b * S + row)) * D + col] = (bf16)(o[d][r] / lrow[r]);
    }
}

// ---------------------------------------------------------------------------
// out = LayerNorm(x + y) * g + b   (one block per token, D=1024)
// ---------------------------------------------------------------------------
__global__ __launch_bounds__(256) void add_ln_kernel(
    const float* __restrict__ x, const float* __restrict__ y,
    const float* __restrict__ g, const float* __restrict__ b,
    float* __restrict__ outf, bf16* __restrict__ outb) {
  const int D = 1024;
  const int row = blockIdx.x, tid = threadIdx.x;
  __shared__ float shs[8], shs2[8];
  __shared__ float shmu, shrs;
  float v[4], s = 0.0f, s2 = 0.0f;
#pragma unroll
  for (int i = 0; i < 4; ++i) {
    int col = tid + i * 256;
    float t = x[(size_t)row * D + col] + y[(size_t)row * D + col];
    v[i] = t; s += t; s2 += t * t;
  }
#pragma unroll
  for (int off = 16; off >= 1; off >>= 1) {
    s += __shfl_xor(s, off);
    s2 += __shfl_xor(s2, off);
  }
  if ((tid & 31) == 0) { shs[tid >> 5] = s; shs2[tid >> 5] = s2; }
  __syncthreads();
  if (tid == 0) {
    float ts = 0.0f, ts2 = 0.0f;
    for (int i = 0; i < 8; ++i) { ts += shs[i]; ts2 += shs2[i]; }
    float mu = ts / D;
    shmu = mu;
    shrs = rsqrtf(ts2 / D - mu * mu + 1e-6f);
  }
  __syncthreads();
  float mu = shmu, rs = shrs;
#pragma unroll
  for (int i = 0; i < 4; ++i) {
    int col = tid + i * 256;
    float o = (v[i] - mu) * rs * g[col] + b[col];
    outf[(size_t)row * D + col] = o;
    if (outb) outb[(size_t)row * D + col] = (bf16)o;
  }
}

// ---------------------------------------------------------------------------
extern "C" void kernel_launch(void* const* d_in, const int* in_sizes, int n_in,
                              void* d_out, int out_size, void* d_ws, size_t ws_size,
                              hipStream_t stream) {
  (void)in_sizes; (void)n_in; (void)out_size; (void)ws_size;
  const int B = 2, S = 2048, D = 1024, DFF = 4096, N = B * S;

  const float* x   = (const float*)d_in[0];
  const float* Wq  = (const float*)d_in[1];
  const float* bq  = (const float*)d_in[2];
  const float* Wk  = (const float*)d_in[3];
  const float* bk  = (const float*)d_in[4];
  const float* Wv  = (const float*)d_in[5];
  const float* bv  = (const float*)d_in[6];
  const float* Wo  = (const float*)d_in[7];
  const float* bo  = (const float*)d_in[8];
  const float* W1  = (const float*)d_in[9];
  const float* b1  = (const float*)d_in[10];
  const float* W2  = (const float*)d_in[11];
  const float* b2  = (const float*)d_in[12];
  const float* g1  = (const float*)d_in[13];
  const float* be1 = (const float*)d_in[14];
  const float* g2  = (const float*)d_in[15];
  const float* be2 = (const float*)d_in[16];
  float* out = (float*)d_out;

  char* wp = (char*)d_ws;
  auto take = [&](size_t bytes) {
    char* p = wp;
    wp += (bytes + 255) & ~(size_t)255;
    return p;
  };
  bf16* xb    = (bf16*)take((size_t)N * D * 2);
  bf16* wqT   = (bf16*)take((size_t)D * D * 2);
  bf16* wkT   = (bf16*)take((size_t)D * D * 2);
  bf16* wvT   = (bf16*)take((size_t)D * D * 2);
  bf16* woT   = (bf16*)take((size_t)D * D * 2);
  bf16* w1T   = (bf16*)take((size_t)D * DFF * 2);   // [DFF][D]
  bf16* w2T   = (bf16*)take((size_t)DFF * D * 2);   // [D][DFF]
  bf16* qb    = (bf16*)take((size_t)N * D * 2);
  bf16* kb    = (bf16*)take((size_t)N * D * 2);
  bf16* vtb   = (bf16*)take((size_t)N * D * 2);     // transposed V: [D][N]
  bf16* ctxb  = (bf16*)take((size_t)N * D * 2);
  float* attnf = (float*)take((size_t)N * D * 4);
  float* x1    = (float*)take((size_t)N * D * 4);
  bf16* x1b   = (bf16*)take((size_t)N * D * 2);
  bf16* hidb  = (bf16*)take((size_t)N * DFF * 2);
  float* ff    = (float*)take((size_t)N * D * 4);

  dim3 blk(256), blkg(128);
  cvt_kernel<<<(N * D + 255) / 256, blk, 0, stream>>>(x, xb, N * D);
  cvt_t_kernel<<<dim3(D / 32, D / 32), blk, 0, stream>>>(Wq, wqT, D, D);
  cvt_t_kernel<<<dim3(D / 32, D / 32), blk, 0, stream>>>(Wk, wkT, D, D);
  cvt_t_kernel<<<dim3(D / 32, D / 32), blk, 0, stream>>>(Wv, wvT, D, D);
  cvt_t_kernel<<<dim3(D / 32, D / 32), blk, 0, stream>>>(Wo, woT, D, D);
  cvt_t_kernel<<<dim3(DFF / 32, D / 32), blk, 0, stream>>>(W1, w1T, D, DFF);
  cvt_t_kernel<<<dim3(D / 32, DFF / 32), blk, 0, stream>>>(W2, w2T, DFF, D);

  // QKV projections (bf16 outputs consumed by attention; V written transposed)
  gemm_bf16_kernel<<<dim3(D / 128, N / 128), blkg, 0, stream>>>(
      xb, wqT, bq, nullptr, qb, N, D, D, 0, 0);
  gemm_bf16_kernel<<<dim3(D / 128, N / 128), blkg, 0, stream>>>(
      xb, wkT, bk, nullptr, kb, N, D, D, 0, 0);
  gemm_bf16_kernel<<<dim3(D / 128, N / 128), blkg, 0, stream>>>(
      xb, wvT, bv, nullptr, vtb, N, D, D, 0, 1);

  attn_kernel<<<dim3(S / 128, B * 16), blk, 0, stream>>>(qb, kb, vtb, ctxb, S, N);

  // output projection + LN1
  gemm_bf16_kernel<<<dim3(D / 128, N / 128), blkg, 0, stream>>>(
      ctxb, woT, bo, attnf, nullptr, N, D, D, 0, 0);
  add_ln_kernel<<<N, blk, 0, stream>>>(x, attnf, g1, be1, x1, x1b);

  // FFN + LN2
  gemm_bf16_kernel<<<dim3(DFF / 128, N / 128), blkg, 0, stream>>>(
      x1b, w1T, b1, nullptr, hidb, N, DFF, D, 1, 0);
  gemm_bf16_kernel<<<dim3(D / 128, N / 128), blkg, 0, stream>>>(
      hidb, w2T, b2, ff, nullptr, N, D, DFF, 0, 0);
  add_ln_kernel<<<N, blk, 0, stream>>>(x1, ff, g2, be2, out, nullptr);
}
